// NeuralODEModel_21157008900131
// MI455X (gfx1250) — matrix operand order repaired and checked by reading, not compiled, not run on hardware
//
#include <hip/hip_runtime.h>
#include <hip/hip_bf16.h>

// ---------------------------------------------------------------------------
// NeuralODE (2-layer LSTM -> RK4 ODE -> FC) for MI455X / gfx1250.
// Matmuls run on V_WMMA_F32_16X16X32_BF16 (f32 accumulate), weights converted
// to bf16 once per call (L2-resident: ~27MB << 192MB L2). LDS tiles are filled
// with GLOBAL_LOAD_ASYNC_TO_LDS_B128 (ASYNCcnt) double-buffered so global
// latency overlaps the WMMA work. Cell state and ODE state stay fp32.
// ---------------------------------------------------------------------------

typedef unsigned short u16;
typedef unsigned int   u32;
typedef __attribute__((ext_vector_type(16))) __bf16 v16bf;
typedef __attribute__((ext_vector_type(8)))  float  v8f;

union Frag { v16bf v; u32 u[8]; };

#define LB   512   // batch
#define LS   256   // seq len
#define LI   32    // input
#define LH   1024  // hidden
#define LG   4096  // 4*H gates
#define LOUT 4

#define BM 128
#define BN 128
#define BK 32
#define ASTR 40    // padded LDS row stride (ushorts): 80B -> 16B-aligned rows, conflict-free

__device__ __forceinline__ u16 f2bf(float f) {
    u32 u = __float_as_uint(f);
    u32 r = (u + 0x7FFFu + ((u >> 16) & 1u)) >> 16;   // RNE
    return (u16)r;
}
__device__ __forceinline__ float sigf(float x) {
    return 1.0f / (1.0f + __expf(-x));
}

// memory -> LDS direct, tracked by ASYNCcnt (gfx1250 VGLOBAL op 98)
__device__ __forceinline__ void async_ld_b128(u32 lds_addr, const void* gaddr) {
    asm volatile("global_load_async_to_lds_b128 %0, %1, off"
                 :: "v"(lds_addr), "v"(gaddr) : "memory");
}
__device__ __forceinline__ void wait_async0() {
    asm volatile("s_wait_asynccnt 0x0" ::: "memory");
}

// ---------------------------------------------------------------------------
// WMMA GEMM: out[M,N] = A1[M,K1] @ W1[N,K1]^T (+ A2[M,K2] @ W2[N,K2]^T) + bias
// A, W are bf16 row-major; out fp32. M fixed = gridDim.y*BM. Block: 256 thr,
// 8 waves in 4(M)x2(N); each wave owns a 32x64 tile = 2x4 WMMA 16x16 tiles.
// Double-buffered LDS, filled asynchronously.
// ---------------------------------------------------------------------------
template <bool DUAL, bool RELU>
__global__ __launch_bounds__(256) void gemm_tiles(
    const u16* __restrict__ A1, const u16* __restrict__ W1, int K1,
    const u16* __restrict__ A2, const u16* __restrict__ W2, int K2,
    const float* __restrict__ bias, float* __restrict__ out, int N) {
    __shared__ u16 At[2][BM * ASTR];
    __shared__ u16 Bt[2][BM * ASTR];

    const int tid   = threadIdx.x;
    const int lane  = tid & 31;
    const int wid   = tid >> 5;
    const int waveM = wid & 3;        // 0..3
    const int waveN = wid >> 2;       // 0..1
    const int lm    = lane & 15;
    const int hf    = lane >> 4;      // lane half
    const int bm    = blockIdx.y * BM;
    const int bn    = blockIdx.x * BN;

    const v8f vz = {};
    v8f acc[2][4];
#pragma unroll
    for (int i = 0; i < 2; ++i)
#pragma unroll
        for (int j = 0; j < 4; ++j) acc[i][j] = vz;

    const int r0  = tid >> 2;   // 0..63 : row loaded by this thread
    const int seg = tid & 3;    // 0..3  : 8-element k segment

    const int nch1 = K1 / BK;
    const int nch  = nch1 + (DUAL ? (K2 / BK) : 0);

    // issue all async tile loads for flattened chunk j into LDS buffer `buf`
    auto issue = [&](int j, int buf) {
        const u16* A;
        const u16* W;
        int kc;
        if (!DUAL || j < nch1) { A = A1; W = W1; kc = j * BK; }
        else                   { A = A2; W = W2; kc = (j - nch1) * BK; }
        const int K = (!DUAL || j < nch1) ? K1 : K2;
#pragma unroll
        for (int rr = 0; rr < BM; rr += 64) {
            const int r = r0 + rr;
            const u16* ga = A + (size_t)(bm + r) * K + kc + seg * 8;
            async_ld_b128((u32)(uintptr_t)&At[buf][r * ASTR + seg * 8], ga);
            const u16* gw = W + (size_t)(bn + r) * K + kc + seg * 8;
            async_ld_b128((u32)(uintptr_t)&Bt[buf][r * ASTR + seg * 8], gw);
        }
    };

    issue(0, 0);
    for (int j = 0; j < nch; ++j) {
        const int buf = j & 1;
        wait_async0();       // my chunk-j loads have landed in LDS
        __syncthreads();     // ... and so have everyone else's
        if (j + 1 < nch) issue(j + 1, buf ^ 1);  // overlaps with WMMA below

        // A fragments: ISA 16-bit 16x32 layout. lane: M=lane%16, half=lane/16
        // vgpr v<4 : K = 2v + 8*half ; v>=4 : K = 16 + 2(v-4) + 8*half
        Frag af[2];
#pragma unroll
        for (int mt = 0; mt < 2; ++mt) {
            const int mrow = (waveM * 32 + mt * 16 + lm) * ASTR;
#pragma unroll
            for (int v = 0; v < 8; ++v) {
                const int k = ((v < 4) ? (2 * v) : (16 + 2 * (v - 4))) + 8 * hf;
                af[mt].u[v] = *(const u32*)&At[buf][mrow + k];
            }
        }
        // B fragments: N=lane%16; same K layout as A
#pragma unroll
        for (int nt = 0; nt < 4; ++nt) {
            Frag bfr;
            const int nrow = (waveN * 64 + nt * 16 + lm) * ASTR;
#pragma unroll
            for (int v = 0; v < 8; ++v) {
                const int k = ((v < 4) ? (2 * v) : (16 + 2 * (v - 4))) + 8 * hf;
                bfr.u[v] = *(const u32*)&Bt[buf][nrow + k];
            }
#pragma unroll
            for (int mt = 0; mt < 2; ++mt) {
                acc[mt][nt] = __builtin_amdgcn_wmma_f32_16x16x32_bf16(
                    false, af[mt].v, false, bfr.v, (short)0, acc[mt][nt],
                    false, false);
            }
        }
        __syncthreads();     // done reading buf before it is refilled next iter
    }

    // Epilogue. D layout: vgpr r -> M = r + 8*(lane/16), N = lane%16.
#pragma unroll
    for (int mt = 0; mt < 2; ++mt) {
#pragma unroll
        for (int nt = 0; nt < 4; ++nt) {
            const int gn = bn + waveN * 64 + nt * 16 + lm;
            const float bv = bias ? bias[gn] : 0.0f;
            const int gm0 = bm + waveM * 32 + mt * 16 + 8 * hf;
#pragma unroll
            for (int r = 0; r < 8; ++r) {
                float v = acc[mt][nt][r] + bv;
                if (RELU) v = fmaxf(v, 0.0f);
                out[(size_t)(gm0 + r) * N + gn] = v;
            }
        }
    }
}

// ---------------------------------------------------------------------------
// Elementwise kernels
// ---------------------------------------------------------------------------
__global__ void cvt_bf16(const float* __restrict__ src, u16* __restrict__ dst, int n) {
    int i = blockIdx.x * blockDim.x + threadIdx.x;
    if (i < n) dst[i] = f2bf(src[i]);
}

// x[B,S,I] fp32 -> xbf[S,B,I] bf16 (time-major for per-step GEMM slices)
__global__ void xpose_x(const float* __restrict__ x, u16* __restrict__ xbf) {
    int idx = blockIdx.x * blockDim.x + threadIdx.x;
    if (idx >= LB * LS * LI) return;
    int i = idx % LI;
    int s = (idx / LI) % LS;
    int b = idx / (LI * LS);
    xbf[((size_t)s * LB + b) * LI + i] = f2bf(x[idx]);
}

__global__ void bias_comb(const float* __restrict__ a0, const float* __restrict__ b0,
                          const float* __restrict__ a1, const float* __restrict__ b1,
                          float* __restrict__ o0, float* __restrict__ o1, int n) {
    int i = blockIdx.x * blockDim.x + threadIdx.x;
    if (i < n) { o0[i] = a0[i] + b0[i]; o1[i] = a1[i] + b1[i]; }
}

__global__ void zero16(uint4* __restrict__ p, long n16) {
    long i = (long)blockIdx.x * blockDim.x + threadIdx.x;
    if (i < n16) { uint4 z = {0, 0, 0, 0}; p[i] = z; }
}

// PyTorch gate order i,f,g,o along the 4H axis of `gates`
__global__ void lstm_update(const float* __restrict__ gates, float* __restrict__ c,
                            u16* __restrict__ hbf, float* __restrict__ hf32, int n) {
    int idx = blockIdx.x * blockDim.x + threadIdx.x;
    if (idx >= n) return;
    int b = idx / LH, hh = idx % LH;
    const float* g = gates + (size_t)b * LG;
    float gi = sigf(g[hh]);
    float gf = sigf(g[LH + hh]);
    float gg = tanhf(g[2 * LH + hh]);
    float go = sigf(g[3 * LH + hh]);
    float cv = gf * c[idx] + gi * gg;
    c[idx] = cv;
    float hv = go * tanhf(cv);
    hbf[idx] = f2bf(hv);
    if (hf32) hf32[idx] = hv;
}

__global__ void axpy_bf(const float* __restrict__ h, const float* __restrict__ k,
                        float a, u16* __restrict__ obf, int n) {
    int i = blockIdx.x * blockDim.x + threadIdx.x;
    if (i < n) obf[i] = f2bf(fmaf(a, k[i], h[i]));
}

__global__ void rk4_combine(float* __restrict__ h, const float* __restrict__ k1,
                            const float* __restrict__ k2, const float* __restrict__ k3,
                            const float* __restrict__ k4, float s,
                            u16* __restrict__ hbf, int n) {
    int i = blockIdx.x * blockDim.x + threadIdx.x;
    if (i < n) {
        float v = h[i] + s * (k1[i] + 2.0f * k2[i] + 2.0f * k3[i] + k4[i]);
        h[i] = v;
        hbf[i] = f2bf(v);
    }
}

// out[b,o] = [h_ode, t_feat] @ W_fc^T + b_fc   (K = H+2, OUT=4 -> scalar dots)
__global__ void fc_head(const float* __restrict__ h, const float* __restrict__ tf,
                        const float* __restrict__ Wfc, const float* __restrict__ bfc,
                        float* __restrict__ out) {
    int idx = blockIdx.x * blockDim.x + threadIdx.x;
    if (idx >= LB * LOUT) return;
    int b = idx >> 2, o = idx & 3;
    const float* w = Wfc + (size_t)o * (LH + 2);
    const float* hb = h + (size_t)b * LH;
    float s = bfc[o];
    for (int k = 0; k < LH; ++k) s = fmaf(hb[k], w[k], s);
    s = fmaf(tf[b * 2 + 0], w[LH + 0], s);
    s = fmaf(tf[b * 2 + 1], w[LH + 1], s);
    out[idx] = s;
}

// ---------------------------------------------------------------------------
// Host orchestration
// ---------------------------------------------------------------------------
extern "C" void kernel_launch(void* const* d_in, const int* in_sizes, int n_in,
                              void* d_out, int out_size, void* d_ws, size_t ws_size,
                              hipStream_t stream) {
    const float* x      = (const float*)d_in[0];
    const float* t_feat = (const float*)d_in[1];
    const float* Wih0   = (const float*)d_in[2];
    const float* Whh0   = (const float*)d_in[3];
    const float* bih0   = (const float*)d_in[4];
    const float* bhh0   = (const float*)d_in[5];
    const float* Wih1   = (const float*)d_in[6];
    const float* Whh1   = (const float*)d_in[7];
    const float* bih1   = (const float*)d_in[8];
    const float* bhh1   = (const float*)d_in[9];
    const float* W_ode  = (const float*)d_in[10];
    const float* b_ode  = (const float*)d_in[11];
    const float* W_fc   = (const float*)d_in[12];
    const float* b_fc   = (const float*)d_in[13];
    float* out = (float*)d_out;

    char* ws = (char*)d_ws;
    size_t off = 0;
    auto alloc = [&](size_t bytes) -> char* {
        char* p = ws + off;
        off = (off + bytes + 255) & ~(size_t)255;
        return p;
    };

    u16*   xbf     = (u16*)alloc((size_t)LS * LB * LI * 2);
    u16*   wih0bf  = (u16*)alloc((size_t)LG * LI * 2);
    u16*   whh0bf  = (u16*)alloc((size_t)LG * LH * 2);
    u16*   wih1bf  = (u16*)alloc((size_t)LG * LH * 2);
    u16*   whh1bf  = (u16*)alloc((size_t)LG * LH * 2);
    u16*   wodebf  = (u16*)alloc((size_t)LH * LH * 2);
    float* b0      = (float*)alloc((size_t)LG * 4);
    float* b1      = (float*)alloc((size_t)LG * 4);
    // zeroed region (contiguous: sizes are multiples of 256)
    u16*   h1bf    = (u16*)alloc((size_t)LB * LH * 2);
    u16*   h2bf    = (u16*)alloc((size_t)LB * LH * 2);
    float* c1      = (float*)alloc((size_t)LB * LH * 4);
    float* c2      = (float*)alloc((size_t)LB * LH * 4);
    // end zeroed region
    float* h2f     = (float*)alloc((size_t)LB * LH * 4);
    u16*   tmpbf   = (u16*)alloc((size_t)LB * LH * 2);
    float* gates   = (float*)alloc((size_t)LB * LG * 4);
    float* k1      = (float*)alloc((size_t)LB * LH * 4);
    float* k2      = (float*)alloc((size_t)LB * LH * 4);
    float* k3      = (float*)alloc((size_t)LB * LH * 4);
    float* k4      = (float*)alloc((size_t)LB * LH * 4);
    (void)ws_size; (void)n_in; (void)in_sizes; (void)out_size;

    const int nBH = LB * LH;                 // 524288
    const int TB = 256;
    auto nb = [](int n, int t) { return (n + t - 1) / t; };

    // ---- prep: weight/bias/x conversion, zero state -------------------------
    cvt_bf16<<<nb(LG * LI, TB), TB, 0, stream>>>(Wih0, wih0bf, LG * LI);
    cvt_bf16<<<nb(LG * LH, TB), TB, 0, stream>>>(Whh0, whh0bf, LG * LH);
    cvt_bf16<<<nb(LG * LH, TB), TB, 0, stream>>>(Wih1, wih1bf, LG * LH);
    cvt_bf16<<<nb(LG * LH, TB), TB, 0, stream>>>(Whh1, whh1bf, LG * LH);
    cvt_bf16<<<nb(LH * LH, TB), TB, 0, stream>>>(W_ode, wodebf, LH * LH);
    bias_comb<<<nb(LG, TB), TB, 0, stream>>>(bih0, bhh0, bih1, bhh1, b0, b1, LG);
    xpose_x<<<nb(LB * LS * LI, TB), TB, 0, stream>>>(x, xbf);
    {
        long zbytes = (long)nBH * 2 * 2 + (long)nBH * 4 * 2;  // h1bf,h2bf,c1,c2
        long n16 = zbytes / 16;
        zero16<<<nb((int)n16, TB), TB, 0, stream>>>((uint4*)h1bf, n16);
    }

    // ---- 2-layer LSTM, fused per-step ---------------------------------------
    const dim3 gGate(LG / BN, LB / BM);      // (32, 4)
    for (int t = 0; t < LS; ++t) {
        const u16* xt = xbf + (size_t)t * LB * LI;
        // layer 0: gates = x_t@Wih0^T + h1@Whh0^T + b0
        gemm_tiles<true, false><<<gGate, 256, 0, stream>>>(
            xt, wih0bf, LI, h1bf, whh0bf, LH, b0, gates, LG);
        lstm_update<<<nb(nBH, TB), TB, 0, stream>>>(gates, c1, h1bf, (float*)nullptr, nBH);
        // layer 1: gates = h1@Wih1^T + h2@Whh1^T + b1
        gemm_tiles<true, false><<<gGate, 256, 0, stream>>>(
            h1bf, wih1bf, LH, h2bf, whh1bf, LH, b1, gates, LG);
        lstm_update<<<nb(nBH, TB), TB, 0, stream>>>(gates, c2, h2bf, h2f, nBH);
    }

    // ---- Neural ODE: RK4, f(h) = relu(h@W_ode^T + b_ode) --------------------
    const dim3 gOde(LH / BN, LB / BM);       // (8, 4)
    const float dt = 1.0f / 32.0f;
    for (int s = 0; s < 32; ++s) {
        gemm_tiles<false, true><<<gOde, 256, 0, stream>>>(
            h2bf, wodebf, LH, (const u16*)nullptr, (const u16*)nullptr, 0,
            b_ode, k1, LH);
        axpy_bf<<<nb(nBH, TB), TB, 0, stream>>>(h2f, k1, 0.5f * dt, tmpbf, nBH);
        gemm_tiles<false, true><<<gOde, 256, 0, stream>>>(
            tmpbf, wodebf, LH, (const u16*)nullptr, (const u16*)nullptr, 0,
            b_ode, k2, LH);
        axpy_bf<<<nb(nBH, TB), TB, 0, stream>>>(h2f, k2, 0.5f * dt, tmpbf, nBH);
        gemm_tiles<false, true><<<gOde, 256, 0, stream>>>(
            tmpbf, wodebf, LH, (const u16*)nullptr, (const u16*)nullptr, 0,
            b_ode, k3, LH);
        axpy_bf<<<nb(nBH, TB), TB, 0, stream>>>(h2f, k3, dt, tmpbf, nBH);
        gemm_tiles<false, true><<<gOde, 256, 0, stream>>>(
            tmpbf, wodebf, LH, (const u16*)nullptr, (const u16*)nullptr, 0,
            b_ode, k4, LH);
        rk4_combine<<<nb(nBH, TB), TB, 0, stream>>>(h2f, k1, k2, k3, k4,
                                                    dt / 6.0f, h2bf, nBH);
    }

    // ---- FC head ------------------------------------------------------------
    fc_head<<<nb(LB * LOUT, TB), TB, 0, stream>>>(h2f, t_feat, W_fc, b_fc, out);
}
